// PointNetSetAbstraction_35974646071888
// MI455X (gfx1250) — matrix operand order repaired and checked by reading, not compiled, not run on hardware
//
#include <hip/hip_runtime.h>
#include <hip/hip_bf16.h>

typedef _Float16 half_t;
typedef __attribute__((ext_vector_type(16))) _Float16 v16h;
typedef __attribute__((ext_vector_type(8)))  float    v8f;

#define B_       16
#define N_       4096
#define S_       1024
#define KNN_     32
#define CPTS_    64
#define MTOT_    (B_ * S_ * KNN_)      // 524288
#define EPS_     1e-5f

union FragH {
    v16h v;
    uint4 q[2];
};

// ---------------------------------------------------------------------------
// Kernel 1: farthest point sampling. One block per batch, 256 threads,
// xyz resident in LDS, sequential 1024-step min-dist/argmax scan.
// Writes centroid indices to ws and new_xyz to d_out[0 .. B*S*3).
// ---------------------------------------------------------------------------
__global__ void fps_kernel(const float* __restrict__ xyz,
                           int* __restrict__ cent_idx,
                           float* __restrict__ out_newxyz) {
    __shared__ __align__(16) float xyz_l[N_ * 3];
    __shared__ float red_val[256];
    __shared__ int   red_idx[256];
    __shared__ int   winner;

    const int b   = blockIdx.x;
    const int tid = threadIdx.x;

    for (int e = tid; e < N_ * 3; e += 256)
        xyz_l[e] = xyz[(size_t)b * N_ * 3 + e];
    __syncthreads();

    // 16 points per thread, strided
    float mind[16];
    #pragma unroll
    for (int j = 0; j < 16; ++j) mind[j] = 1e10f;

    int farthest = 0;
    for (int s = 0; s < S_; ++s) {
        // record current farthest as centroid s
        if (tid == 0) {
            cent_idx[b * S_ + s] = farthest;
            out_newxyz[((size_t)b * S_ + s) * 3 + 0] = xyz_l[farthest * 3 + 0];
            out_newxyz[((size_t)b * S_ + s) * 3 + 1] = xyz_l[farthest * 3 + 1];
            out_newxyz[((size_t)b * S_ + s) * 3 + 2] = xyz_l[farthest * 3 + 2];
        }
        const float cx = xyz_l[farthest * 3 + 0];
        const float cy = xyz_l[farthest * 3 + 1];
        const float cz = xyz_l[farthest * 3 + 2];

        float bestv = -1.0f;
        int   besti = 0;
        #pragma unroll
        for (int j = 0; j < 16; ++j) {
            const int p = tid + j * 256;
            const float dx = xyz_l[p * 3 + 0] - cx;
            const float dy = xyz_l[p * 3 + 1] - cy;
            const float dz = xyz_l[p * 3 + 2] - cz;
            const float d  = dx * dx + dy * dy + dz * dz;
            mind[j] = fminf(mind[j], d);
            if (mind[j] > bestv) { bestv = mind[j]; besti = p; }
        }
        red_val[tid] = bestv;
        red_idx[tid] = besti;
        __syncthreads();
        for (int h = 128; h > 0; h >>= 1) {
            if (tid < h) {
                if (red_val[tid + h] > red_val[tid]) {
                    red_val[tid] = red_val[tid + h];
                    red_idx[tid] = red_idx[tid + h];
                }
            }
            __syncthreads();
        }
        if (tid == 0) winner = red_idx[0];
        __syncthreads();
        farthest = winner;
        __syncthreads();
    }
}

// ---------------------------------------------------------------------------
// Kernel 2: KNN (top-32 nearest) per centroid. One block per (b,s),
// 128 threads. 4096 squared distances in LDS, 32 argmin rounds.
// ---------------------------------------------------------------------------
__global__ void knn_kernel(const float* __restrict__ xyz,
                           const float* __restrict__ newxyz,
                           int* __restrict__ knn_idx) {
    __shared__ __align__(16) float dist[N_];
    __shared__ float red_val[128];
    __shared__ int   red_idx[128];

    const int bs  = blockIdx.x;          // b*S_ + s
    const int b   = bs >> 10;
    const int tid = threadIdx.x;

    const float cx = newxyz[(size_t)bs * 3 + 0];
    const float cy = newxyz[(size_t)bs * 3 + 1];
    const float cz = newxyz[(size_t)bs * 3 + 2];

    const float* xb = xyz + (size_t)b * N_ * 3;
    for (int j = 0; j < N_ / 128; ++j) {
        const int n = tid + j * 128;
        const float dx = xb[n * 3 + 0] - cx;
        const float dy = xb[n * 3 + 1] - cy;
        const float dz = xb[n * 3 + 2] - cz;
        dist[n] = dx * dx + dy * dy + dz * dz;
    }
    __syncthreads();

    for (int r = 0; r < KNN_; ++r) {
        float bestv = 3.4e38f;
        int   besti = 0;
        for (int j = 0; j < N_ / 128; ++j) {
            const int n = tid + j * 128;
            const float d = dist[n];
            if (d < bestv) { bestv = d; besti = n; }
        }
        red_val[tid] = bestv;
        red_idx[tid] = besti;
        __syncthreads();
        for (int h = 64; h > 0; h >>= 1) {
            if (tid < h) {
                if (red_val[tid + h] < red_val[tid]) {
                    red_val[tid] = red_val[tid + h];
                    red_idx[tid] = red_idx[tid + h];
                }
            }
            __syncthreads();
        }
        if (tid == 0) {
            knn_idx[(size_t)bs * KNN_ + r] = red_idx[0];
            dist[red_idx[0]] = 3.4e38f;
        }
        __syncthreads();
    }
}

// ---------------------------------------------------------------------------
// Kernel 3: zero stats buffer
// ---------------------------------------------------------------------------
__global__ void zero_kernel(float* p, int n) {
    const int t = blockIdx.x * blockDim.x + threadIdx.x;
    if (t < n) p[t] = 0.0f;
}

// ---------------------------------------------------------------------------
// WMMA fragment loaders (documented CDNA5 16-bit layouts).
// A 16x32 (MxK): lane L holds row M=L&15; halves h=0..7 -> K = kb+off8+h,
//                h=8..15 -> K = kb+16+off8+(h-8), off8 = (L>>4)*8.
// B 32x16 (KxN): lane L holds col N=L&15; same K mapping as A.
// ---------------------------------------------------------------------------
__device__ inline v16h load_a_frag(const half_t* rowbase, int kb, int lane) {
    FragH f;
    const int off8 = (lane >> 4) << 3;
    f.q[0] = *(const uint4*)(rowbase + kb + off8);
    f.q[1] = *(const uint4*)(rowbase + kb + 16 + off8);
    return f.v;
}
__device__ inline v16h load_b_frag(const half_t* wl, int cinp, int ncol, int kb, int lane) {
    FragH f;
    const half_t* p = wl + (size_t)ncol * cinp + kb + ((lane >> 4) << 3);
    f.q[0] = *(const uint4*)(p);
    f.q[1] = *(const uint4*)(p + 16);
    return f.v;
}

// ---------------------------------------------------------------------------
// Kernel 4: layer-1 GEMM with fused gather (xyz-diff ++ points), f16 WMMA,
// f32 accumulate, +bias, store Y1 as f16.  Cin=67 (padded 96), Cout=64.
// Block: 256 threads = 8 waves; M-tile = 128 rows.
// ---------------------------------------------------------------------------
__global__ void gemm1_gather_kernel(const float* __restrict__ xyz,
                                    const float* __restrict__ points,
                                    const int*   __restrict__ knn_idx,
                                    const float* __restrict__ newxyz,
                                    const float* __restrict__ w1,
                                    const float* __restrict__ b1,
                                    half_t* __restrict__ Y1) {
    constexpr int CIN = 67, CINP = 96, COUT = 64, NT = 4, KS = 3;
    __shared__ __align__(16) half_t Al[128 * CINP];
    __shared__ __align__(16) half_t Wl[COUT * CINP];

    const int tid = threadIdx.x;
    const long rowBase = (long)blockIdx.x * 128;

    for (int e = tid; e < COUT * CINP; e += 256) {
        const int o = e / CINP, c = e % CINP;
        Wl[e] = (c < CIN) ? (half_t)w1[o * CIN + c] : (half_t)0.0f;
    }
    for (int e = tid; e < 128 * CINP; e += 256) {
        const int i = e / CINP, c = e % CINP;
        const long m = rowBase + i;
        const int b = (int)(m >> 15);
        const int rem = (int)(m & 32767);
        const int s = rem >> 5;
        float v = 0.0f;
        const int pidx = knn_idx[m];
        if (c < 3) {
            v = xyz[((size_t)b * N_ + pidx) * 3 + c] - newxyz[((size_t)b * S_ + s) * 3 + c];
        } else if (c < CIN) {
            v = points[((size_t)b * N_ + pidx) * CPTS_ + (c - 3)];
        }
        Al[e] = (half_t)v;
    }
    __syncthreads();

    const int wave = tid >> 5;
    const int lane = tid & 31;
    const half_t* arow = &Al[(wave * 16 + (lane & 15)) * CINP];

    for (int nt = 0; nt < NT; ++nt) {
        v8f acc = {};
        #pragma unroll
        for (int ks = 0; ks < KS; ++ks) {
            v16h a = load_a_frag(arow, ks * 32, lane);
            v16h b = load_b_frag(Wl, CINP, nt * 16 + (lane & 15), ks * 32, lane);
            acc = __builtin_amdgcn_wmma_f32_16x16x32_f16(false, a, false, b,
                                                         (short)0, acc, false, false);
        }
        const int col = nt * 16 + (lane & 15);
        const float bias = b1[col];
        #pragma unroll
        for (int r = 0; r < 8; ++r) {
            const long row = rowBase + wave * 16 + r + ((lane >> 4) << 3);
            Y1[row * COUT + col] = (half_t)(acc[r] + bias);
        }
    }
}

// ---------------------------------------------------------------------------
// Kernel 5/7: GEMM with fused previous-layer BN(training stats)+ReLU on the
// A operand.  Cin=64, Cout = 64 or 128.  f16 WMMA, f32 accumulate.
// ---------------------------------------------------------------------------
template <int COUT>
__global__ void gemm_bn_kernel(const half_t* __restrict__ Yin,
                               const float* __restrict__ w,
                               const float* __restrict__ bias,
                               const float* __restrict__ gam,
                               const float* __restrict__ bet,
                               const float* __restrict__ sumArr,
                               const float* __restrict__ sqArr,
                               half_t* __restrict__ Yout) {
    constexpr int CINP = 64, KS = 2, NT = COUT / 16;
    __shared__ __align__(16) half_t Al[128 * CINP];
    __shared__ __align__(16) half_t Wl[COUT * CINP];
    __shared__ float a_scale[CINP];
    __shared__ float a_shift[CINP];

    const int tid = threadIdx.x;
    const long rowBase = (long)blockIdx.x * 128;
    const float inv_n = 1.0f / (float)MTOT_;

    if (tid < CINP) {
        const float mu  = sumArr[tid] * inv_n;
        const float var = sqArr[tid] * inv_n - mu * mu;
        const float istd = rsqrtf(var + EPS_);
        const float sc = istd * gam[tid];
        a_scale[tid] = sc;
        a_shift[tid] = bet[tid] - mu * sc;
    }
    for (int e = tid; e < COUT * CINP; e += 256)
        Wl[e] = (half_t)w[e];
    __syncthreads();

    const half_t* src = Yin + rowBase * CINP;
    __builtin_prefetch(src + (size_t)tid * 32, 0, 1);   // global_prefetch_b8
    for (int e = tid; e < 128 * CINP; e += 256) {
        const int c = e & (CINP - 1);
        float v = (float)src[e] * a_scale[c] + a_shift[c];
        v = fmaxf(v, 0.0f);
        Al[e] = (half_t)v;
    }
    __syncthreads();

    const int wave = tid >> 5;
    const int lane = tid & 31;
    const half_t* arow = &Al[(wave * 16 + (lane & 15)) * CINP];

    for (int nt = 0; nt < NT; ++nt) {
        v8f acc = {};
        #pragma unroll
        for (int ks = 0; ks < KS; ++ks) {
            v16h a = load_a_frag(arow, ks * 32, lane);
            v16h b = load_b_frag(Wl, CINP, nt * 16 + (lane & 15), ks * 32, lane);
            acc = __builtin_amdgcn_wmma_f32_16x16x32_f16(false, a, false, b,
                                                         (short)0, acc, false, false);
        }
        const int col = nt * 16 + (lane & 15);
        const float bs = bias[col];
        #pragma unroll
        for (int r = 0; r < 8; ++r) {
            const long row = rowBase + wave * 16 + r + ((lane >> 4) << 3);
            Yout[row * COUT + col] = (half_t)(acc[r] + bs);
        }
    }
}

// ---------------------------------------------------------------------------
// Kernel 6/8: per-channel sum / sum-of-squares for BatchNorm training stats.
// Block-level LDS reduction, then per-channel atomics.
// ---------------------------------------------------------------------------
template <int C>
__global__ void stats_kernel(const half_t* __restrict__ Y,
                             float* __restrict__ sumArr,
                             float* __restrict__ sqArr) {
    __shared__ float s1[256];
    __shared__ float s2[256];
    const int tid = threadIdx.x;
    const int c   = tid & (C - 1);
    const int sub = tid / C;
    constexpr int GROUPS = 256 / C;

    const long rowsPerBlock = (MTOT_ + gridDim.x - 1) / gridDim.x;
    const long r0 = (long)blockIdx.x * rowsPerBlock;
    long r1 = r0 + rowsPerBlock;
    if (r1 > MTOT_) r1 = MTOT_;

    float acc = 0.0f, acc2 = 0.0f;
    for (long r = r0 + sub; r < r1; r += GROUPS) {
        const float v = (float)Y[r * C + c];
        acc  += v;
        acc2 += v * v;
    }
    s1[tid] = acc;
    s2[tid] = acc2;
    __syncthreads();
    for (int h = 128; h >= C; h >>= 1) {
        if (tid < h) {
            s1[tid] += s1[tid + h];
            s2[tid] += s2[tid + h];
        }
        __syncthreads();
    }
    if (tid < C) {
        atomicAdd(&sumArr[tid], s1[tid]);
        atomicAdd(&sqArr[tid],  s2[tid]);
    }
}

// ---------------------------------------------------------------------------
// Kernel 9: BN3 + ReLU + max-pool over the 32 neighbors -> new_points.
// ---------------------------------------------------------------------------
__global__ void bn_maxpool_kernel(const half_t* __restrict__ Y3,
                                  const float* __restrict__ gam,
                                  const float* __restrict__ bet,
                                  const float* __restrict__ sumArr,
                                  const float* __restrict__ sqArr,
                                  float* __restrict__ out) {
    const long t = (long)blockIdx.x * blockDim.x + threadIdx.x;
    const long total = (long)B_ * S_ * 128;
    if (t >= total) return;
    const int  c  = (int)(t & 127);
    const long bs = t >> 7;

    const float inv_n = 1.0f / (float)MTOT_;
    const float mu  = sumArr[c] * inv_n;
    const float var = sqArr[c] * inv_n - mu * mu;
    const float istd = rsqrtf(var + EPS_);
    const float sc = istd * gam[c];
    const float sh = bet[c] - mu * sc;

    const half_t* base = Y3 + bs * KNN_ * 128 + c;
    float m = -3.4e38f;
    #pragma unroll 4
    for (int k = 0; k < KNN_; ++k) {
        float v = (float)base[k * 128] * sc + sh;
        v = fmaxf(v, 0.0f);
        m = fmaxf(m, v);
    }
    out[bs * 128 + c] = m;
}

// ---------------------------------------------------------------------------
// Host launcher
// ---------------------------------------------------------------------------
extern "C" void kernel_launch(void* const* d_in, const int* in_sizes, int n_in,
                              void* d_out, int out_size, void* d_ws, size_t ws_size,
                              hipStream_t stream) {
    const float* xyz    = (const float*)d_in[0];
    const float* points = (const float*)d_in[1];
    const float* w1 = (const float*)d_in[2];
    const float* b1 = (const float*)d_in[3];
    const float* g1 = (const float*)d_in[4];
    const float* be1 = (const float*)d_in[5];
    const float* w2 = (const float*)d_in[6];
    const float* b2 = (const float*)d_in[7];
    const float* g2 = (const float*)d_in[8];
    const float* be2 = (const float*)d_in[9];
    const float* w3 = (const float*)d_in[10];
    const float* b3 = (const float*)d_in[11];
    const float* g3 = (const float*)d_in[12];
    const float* be3 = (const float*)d_in[13];

    float* out_newxyz = (float*)d_out;                       // B*S*3
    float* out_newpts = (float*)d_out + (size_t)B_ * S_ * 3; // B*S*128

    // workspace layout (bytes)
    char* ws = (char*)d_ws;
    int*   cent_idx = (int*)(ws);                                   // 64 KB
    int*   knn_idx  = (int*)(ws + 65536);                           // 2 MB
    float* stats    = (float*)(ws + 65536 + 2097152);               // 512 floats
    float* sum1 = stats;        float* sq1 = stats + 64;
    float* sum2 = stats + 128;  float* sq2 = stats + 192;
    float* sum3 = stats + 256;  float* sq3 = stats + 384;
    size_t off = 65536 + 2097152 + 2048;
    off = (off + 255) & ~(size_t)255;
    half_t* YA = (half_t*)(ws + off);                       // 128ch region: Y1 then Y3
    half_t* Y2 = (half_t*)(ws + off + (size_t)MTOT_ * 128 * sizeof(half_t));
    half_t* Y1 = YA;
    half_t* Y3 = YA;

    fps_kernel<<<B_, 256, 0, stream>>>(xyz, cent_idx, out_newxyz);
    knn_kernel<<<B_ * S_, 128, 0, stream>>>(xyz, out_newxyz, knn_idx);
    zero_kernel<<<2, 256, 0, stream>>>(stats, 512);

    gemm1_gather_kernel<<<MTOT_ / 128, 256, 0, stream>>>(
        xyz, points, knn_idx, out_newxyz, w1, b1, Y1);
    stats_kernel<64><<<512, 256, 0, stream>>>(Y1, sum1, sq1);

    gemm_bn_kernel<64><<<MTOT_ / 128, 256, 0, stream>>>(
        Y1, w2, b2, g1, be1, sum1, sq1, Y2);
    stats_kernel<64><<<512, 256, 0, stream>>>(Y2, sum2, sq2);

    gemm_bn_kernel<128><<<MTOT_ / 128, 256, 0, stream>>>(
        Y2, w3, b3, g2, be2, sum2, sq2, Y3);
    stats_kernel<128><<<512, 256, 0, stream>>>(Y3, sum3, sq3);

    bn_maxpool_kernel<<<(B_ * S_ * 128 + 255) / 256, 256, 0, stream>>>(
        Y3, g3, be3, sum3, sq3, out_newpts);
}
